// SelfAttention_12567074308755
// MI455X (gfx1250) — compile-verified, matrix operand order checked by reading, not run on hardware
//
#include <hip/hip_runtime.h>
#include <hip/hip_bf16.h>

// ---------------- problem constants ----------------
#define T_DIM 2048
#define B_DIM 2
#define E_DIM 1024
#define H_DIM 16
#define DH    64
#define ROWS  (T_DIM * B_DIM)   // 4096

typedef __bf16 bf16_t;
typedef __attribute__((ext_vector_type(16))) __bf16 v16bf;
typedef __attribute__((ext_vector_type(8)))  float  v8f;

// Load a 16x32 bf16 WMMA fragment (A-layout; B-layout is symmetric when the
// operand is stored N-major x K).  Per-lane data: row m = lane&15,
// K in [8g, 8g+8) and [16+8g, 16+8g+8)  (g = lane>>4) -> two 16B loads.
__device__ __forceinline__ v16bf load_frag(const bf16_t* base, int stride, int k0) {
    const int lane = threadIdx.x & 31;
    const int g = lane >> 4, m = lane & 15;
    const bf16_t* p = base + (size_t)m * stride + k0 + g * 8;
    union { v16bf v; uint4 u[2]; } f;
    f.u[0] = *(const uint4*)(p);
    f.u[1] = *(const uint4*)(p + 16);
    return f.v;
}

// Warm L2 ahead of the register pipeline (lowers to global_prefetch_b8).
__device__ __forceinline__ void prefetch_frag(const bf16_t* base, int stride, int k0) {
    const int lane = threadIdx.x & 31;
    const int g = lane >> 4, m = lane & 15;
    __builtin_prefetch(base + (size_t)m * stride + k0 + g * 8, 0, 3);
}

// ---------------- fp32 -> bf16 convert ----------------
__global__ void cvt_f32_bf16(const float* __restrict__ in, bf16_t* __restrict__ out, int n) {
    int i = blockIdx.x * blockDim.x + threadIdx.x;
    if (i < n) out[i] = (bf16_t)in[i];
}

// ---------------- transpose weight (E x E) to bf16, Wt[n][k] = W[k][n] ----------------
__global__ __launch_bounds__(256) void transpose_w_bf16(const float* __restrict__ W,
                                                        bf16_t* __restrict__ Wt, int n) {
    __shared__ float tile[32][33];
    const int bx = blockIdx.x * 32, by = blockIdx.y * 32;
    const int tx = threadIdx.x, ty = threadIdx.y;  // block (32,8)
    #pragma unroll
    for (int i = ty; i < 32; i += 8)
        tile[i][tx] = W[(size_t)(by + i) * n + bx + tx];
    __syncthreads();
    #pragma unroll
    for (int i = ty; i < 32; i += 8)
        Wt[(size_t)(bx + i) * n + by + tx] = (bf16_t)tile[tx][i];
}

// ---------------- generic WMMA GEMM: C[M,N] = A[M,K] @ Bt[N,K]^T ----------------
// 32x64 output tile per wave; software-pipelined (register double buffering).
// mode 0: bf16 row-major out (+bias, +optional fp32 residual)
// mode 1: bf16 head-major out  ((b*H+h)*T + t)*DH + d          (Q/K; Q uses scale)
// mode 3: bf16 transposed head ((b*H+h)*DH + d)*T + t          (V)
// mode 4: fp32 row-major out (+bias)                           (final)
__global__ __launch_bounds__(256) void gemm_wmma(
    const bf16_t* __restrict__ A, int lda,
    const bf16_t* __restrict__ Bt,
    const float* __restrict__ bias,
    const float* __restrict__ resid,
    bf16_t* __restrict__ outb, float* __restrict__ outf,
    int M, int N, int K, float scale, int mode)
{
    const int wid = (blockIdx.x * blockDim.x + threadIdx.x) >> 5;
    const int tiles_n = N >> 6;
    const int mt = wid / tiles_n, nt = wid % tiles_n;
    const int row0 = mt * 32, col0 = nt * 64;
    if (row0 >= M) return;

    v8f acc[8];
    #pragma unroll
    for (int t = 0; t < 8; ++t)
        #pragma unroll
        for (int r = 0; r < 8; ++r) acc[t][r] = 0.0f;

    const bf16_t* A0 = A + (size_t)row0 * lda;
    const bf16_t* A1 = A0 + (size_t)16 * lda;
    const bf16_t* B0 = Bt + (size_t)(col0 +  0) * K;
    const bf16_t* B1 = Bt + (size_t)(col0 + 16) * K;
    const bf16_t* B2 = Bt + (size_t)(col0 + 32) * K;
    const bf16_t* B3 = Bt + (size_t)(col0 + 48) * K;

    // pipeline prologue
    v16bf a0 = load_frag(A0, lda, 0);
    v16bf a1 = load_frag(A1, lda, 0);
    v16bf b0 = load_frag(B0, K, 0);
    v16bf b1 = load_frag(B1, K, 0);
    v16bf b2 = load_frag(B2, K, 0);
    v16bf b3 = load_frag(B3, K, 0);

    for (int k0 = 0; k0 < K - 32; k0 += 32) {
        const int kn = k0 + 32;
        // issue next-step loads first; WMMAs below consume the previous regs,
        // so the loads stay in flight across the whole WMMA pack.
        v16bf a0n = load_frag(A0, lda, kn);
        v16bf a1n = load_frag(A1, lda, kn);
        v16bf b0n = load_frag(B0, K, kn);
        v16bf b1n = load_frag(B1, K, kn);
        v16bf b2n = load_frag(B2, K, kn);
        v16bf b3n = load_frag(B3, K, kn);
        // warm L2 two steps ahead (speculative: OOB prefetch is dropped)
        prefetch_frag(B0, K, kn + 32);
        prefetch_frag(B1, K, kn + 32);
        prefetch_frag(B2, K, kn + 32);
        prefetch_frag(B3, K, kn + 32);

        acc[0] = __builtin_amdgcn_wmma_f32_16x16x32_bf16(false, a0, false, b0, (short)0, acc[0], false, false);
        acc[1] = __builtin_amdgcn_wmma_f32_16x16x32_bf16(false, a0, false, b1, (short)0, acc[1], false, false);
        acc[2] = __builtin_amdgcn_wmma_f32_16x16x32_bf16(false, a0, false, b2, (short)0, acc[2], false, false);
        acc[3] = __builtin_amdgcn_wmma_f32_16x16x32_bf16(false, a0, false, b3, (short)0, acc[3], false, false);
        acc[4] = __builtin_amdgcn_wmma_f32_16x16x32_bf16(false, a1, false, b0, (short)0, acc[4], false, false);
        acc[5] = __builtin_amdgcn_wmma_f32_16x16x32_bf16(false, a1, false, b1, (short)0, acc[5], false, false);
        acc[6] = __builtin_amdgcn_wmma_f32_16x16x32_bf16(false, a1, false, b2, (short)0, acc[6], false, false);
        acc[7] = __builtin_amdgcn_wmma_f32_16x16x32_bf16(false, a1, false, b3, (short)0, acc[7], false, false);

        a0 = a0n; a1 = a1n; b0 = b0n; b1 = b1n; b2 = b2n; b3 = b3n;
    }
    // epilogue: last k-step
    acc[0] = __builtin_amdgcn_wmma_f32_16x16x32_bf16(false, a0, false, b0, (short)0, acc[0], false, false);
    acc[1] = __builtin_amdgcn_wmma_f32_16x16x32_bf16(false, a0, false, b1, (short)0, acc[1], false, false);
    acc[2] = __builtin_amdgcn_wmma_f32_16x16x32_bf16(false, a0, false, b2, (short)0, acc[2], false, false);
    acc[3] = __builtin_amdgcn_wmma_f32_16x16x32_bf16(false, a0, false, b3, (short)0, acc[3], false, false);
    acc[4] = __builtin_amdgcn_wmma_f32_16x16x32_bf16(false, a1, false, b0, (short)0, acc[4], false, false);
    acc[5] = __builtin_amdgcn_wmma_f32_16x16x32_bf16(false, a1, false, b1, (short)0, acc[5], false, false);
    acc[6] = __builtin_amdgcn_wmma_f32_16x16x32_bf16(false, a1, false, b2, (short)0, acc[6], false, false);
    acc[7] = __builtin_amdgcn_wmma_f32_16x16x32_bf16(false, a1, false, b3, (short)0, acc[7], false, false);

    const int lane = threadIdx.x & 31;
    const int g = lane >> 4, nl = lane & 15;
    #pragma unroll
    for (int half = 0; half < 2; ++half) {
        #pragma unroll
        for (int t = 0; t < 4; ++t) {
            const int col = col0 + t * 16 + nl;
            const float bv = bias[col];
            #pragma unroll
            for (int r = 0; r < 8; ++r) {
                const int row = row0 + half * 16 + r + 8 * g;
                float v = (acc[half * 4 + t][r] + bv) * scale;
                if (resid) v += resid[(size_t)row * N + col];
                if (mode == 0) {
                    outb[(size_t)row * N + col] = (bf16_t)v;
                } else if (mode == 1) {
                    const int tt = row / B_DIM, bb = row % B_DIM;
                    const int h = col >> 6, d = col & 63;
                    outb[((size_t)(bb * H_DIM + h) * T_DIM + tt) * DH + d] = (bf16_t)v;
                } else if (mode == 3) {
                    const int tt = row / B_DIM, bb = row % B_DIM;
                    const int h = col >> 6, d = col & 63;
                    outb[((size_t)(bb * H_DIM + h) * DH + d) * T_DIM + tt] = (bf16_t)v;
                } else {
                    outf[(size_t)row * N + col] = v;
                }
            }
        }
    }
}

// ---------------- flash attention (strictly causal + zero-logit sink) ----------------
// qh, kh: (BH, T, DH) bf16 row-major.  vt: (BH, DH, T) bf16 (pre-transposed).
// One wave handles one (head, 16-query tile).  Online softmax initialized with
// the sink logit 0:  m=0, l=1, O=0.
__global__ __launch_bounds__(256) void attn_fwd(
    const bf16_t* __restrict__ qh, const bf16_t* __restrict__ kh,
    const bf16_t* __restrict__ vt, bf16_t* __restrict__ attnb)
{
    __shared__ __attribute__((aligned(16))) bf16_t pbuf[8][16 * 32];
    const int warp = threadIdx.x >> 5;
    const int lane = threadIdx.x & 31;
    const int g = lane >> 4, nl = lane & 15;
    const int wid = blockIdx.x * 8 + warp;
    const int head = wid >> 7;     // 32 heads
    const int qt = wid & 127;      // 128 query tiles
    const int i0 = qt * 16;

    const bf16_t* qbase = qh + ((size_t)head * T_DIM + i0) * DH;
    v16bf qf0 = load_frag(qbase, DH, 0);
    v16bf qf1 = load_frag(qbase, DH, 32);

    v8f m_run, l_run, acc[4];
    #pragma unroll
    for (int r = 0; r < 8; ++r) { m_run[r] = 0.0f; l_run[r] = 1.0f; }
    #pragma unroll
    for (int t = 0; t < 4; ++t)
        #pragma unroll
        for (int r = 0; r < 8; ++r) acc[t][r] = 0.0f;

    bf16_t* pl = &pbuf[warp][0];

    for (int j0 = 0; j0 < i0 + 16; j0 += 32) {
        v8f s[2];
        #pragma unroll
        for (int sub = 0; sub < 2; ++sub) {
            v8f z;
            #pragma unroll
            for (int r = 0; r < 8; ++r) z[r] = 0.0f;
            const int jbase = j0 + sub * 16;
            if (jbase < T_DIM) {   // wave-uniform: EXEC stays full for WMMA
                const bf16_t* kb = kh + ((size_t)head * T_DIM + jbase) * DH;
                v16bf kf0 = load_frag(kb, DH, 0);
                v16bf kf1 = load_frag(kb, DH, 32);
                z = __builtin_amdgcn_wmma_f32_16x16x32_bf16(
                    false, qf0, false, kf0, (short)0, z, false, false);
                z = __builtin_amdgcn_wmma_f32_16x16x32_bf16(
                    false, qf1, false, kf1, (short)0, z, false, false);
            }
            const int j = jbase + nl;
            #pragma unroll
            for (int r = 0; r < 8; ++r) {
                const int irow = i0 + r + 8 * g;
                if (j >= irow || jbase >= T_DIM) z[r] = -1.0e30f;  // strictly causal
            }
            s[sub] = z;
        }

        // V fragments are independent of the softmax -> load them now so the
        // global loads overlap the exp/shuffle VALU work below.
        v16bf vf[4];
        #pragma unroll
        for (int dt = 0; dt < 4; ++dt) {
            const bf16_t* vb = vt + ((size_t)head * DH + dt * 16) * T_DIM + j0;
            vf[dt] = load_frag(vb, T_DIM, 0);
        }

        // online softmax update (row stats live per lane for rows r+8g)
        v8f alpha, p0, p1;
        #pragma unroll
        for (int r = 0; r < 8; ++r) {
            float rm = fmaxf(s[0][r], s[1][r]);
            rm = fmaxf(rm, __shfl_xor(rm, 1));
            rm = fmaxf(rm, __shfl_xor(rm, 2));
            rm = fmaxf(rm, __shfl_xor(rm, 4));
            rm = fmaxf(rm, __shfl_xor(rm, 8));
            const float mn = fmaxf(m_run[r], rm);
            alpha[r] = __expf(m_run[r] - mn);
            p0[r] = __expf(s[0][r] - mn);
            p1[r] = __expf(s[1][r] - mn);
            float rs = p0[r] + p1[r];
            rs += __shfl_xor(rs, 1);
            rs += __shfl_xor(rs, 2);
            rs += __shfl_xor(rs, 4);
            rs += __shfl_xor(rs, 8);
            l_run[r] = l_run[r] * alpha[r] + rs;
            m_run[r] = mn;
        }
        #pragma unroll
        for (int t = 0; t < 4; ++t)
            #pragma unroll
            for (int r = 0; r < 8; ++r) acc[t][r] *= alpha[r];

        // C-layout P -> row-major 16x32 bf16 in LDS, then reload as A-fragment
        #pragma unroll
        for (int r = 0; r < 8; ++r) {
            pl[(r + 8 * g) * 32 + nl]      = (bf16_t)p0[r];
            pl[(r + 8 * g) * 32 + 16 + nl] = (bf16_t)p1[r];
        }
        asm volatile("s_wait_dscnt 0" ::: "memory");   // intra-wave LDS visibility
        v16bf pf = load_frag(pl, 32, 0);

        // O += P @ V   (V stored DH-major so it is already the "Bt" operand)
        #pragma unroll
        for (int dt = 0; dt < 4; ++dt) {
            acc[dt] = __builtin_amdgcn_wmma_f32_16x16x32_bf16(
                false, pf, false, vf[dt], (short)0, acc[dt], false, false);
        }
    }

    // finalize: O /= l, scatter back to (T, B, E) layout as bf16
    const int bb = head / H_DIM, h = head % H_DIM;
    #pragma unroll
    for (int dt = 0; dt < 4; ++dt) {
        #pragma unroll
        for (int r = 0; r < 8; ++r) {
            const int trow = i0 + r + 8 * g;
            const float v = acc[dt][r] / l_run[r];
            attnb[((size_t)trow * B_DIM + bb) * E_DIM + h * DH + dt * 16 + nl] = (bf16_t)v;
        }
    }
}

// ---------------- host-side launch ----------------
extern "C" void kernel_launch(void* const* d_in, const int* in_sizes, int n_in,
                              void* d_out, int out_size, void* d_ws, size_t ws_size,
                              hipStream_t stream) {
    const float* x   = (const float*)d_in[0];
    const float* Wq  = (const float*)d_in[1];
    const float* bq  = (const float*)d_in[2];
    const float* Wk  = (const float*)d_in[3];
    const float* bk  = (const float*)d_in[4];
    const float* Wv  = (const float*)d_in[5];
    const float* bv  = (const float*)d_in[6];
    const float* Wo  = (const float*)d_in[7];
    const float* bo  = (const float*)d_in[8];
    const float* Wln = (const float*)d_in[9];
    const float* bln = (const float*)d_in[10];
    float* out = (float*)d_out;

    char* ws = (char*)d_ws;
    size_t off = 0;
    auto walloc = [&](size_t bytes) {
        char* p = ws + off;
        off = (off + bytes + 4095) & ~(size_t)4095;   // 4KB pad between buffers
        return p;
    };
    bf16_t* xb    = (bf16_t*)walloc((size_t)ROWS * E_DIM * 2);
    bf16_t* wqt   = (bf16_t*)walloc((size_t)E_DIM * E_DIM * 2);
    bf16_t* wkt   = (bf16_t*)walloc((size_t)E_DIM * E_DIM * 2);
    bf16_t* wvt   = (bf16_t*)walloc((size_t)E_DIM * E_DIM * 2);
    bf16_t* wot   = (bf16_t*)walloc((size_t)E_DIM * E_DIM * 2);
    bf16_t* wlnt  = (bf16_t*)walloc((size_t)E_DIM * E_DIM * 2);
    bf16_t* qhb   = (bf16_t*)walloc((size_t)B_DIM * H_DIM * T_DIM * DH * 2);
    bf16_t* khb   = (bf16_t*)walloc((size_t)B_DIM * H_DIM * T_DIM * DH * 2);
    bf16_t* vtb   = (bf16_t*)walloc((size_t)B_DIM * H_DIM * T_DIM * DH * 2);
    bf16_t* attnb = (bf16_t*)walloc((size_t)ROWS * E_DIM * 2);
    bf16_t* yb    = (bf16_t*)walloc((size_t)ROWS * E_DIM * 2);

    const int n_x = ROWS * E_DIM;
    cvt_f32_bf16<<<(n_x + 255) / 256, 256, 0, stream>>>(x, xb, n_x);

    dim3 tb(32, 8), tg(E_DIM / 32, E_DIM / 32);
    transpose_w_bf16<<<tg, tb, 0, stream>>>(Wq,  wqt,  E_DIM);
    transpose_w_bf16<<<tg, tb, 0, stream>>>(Wk,  wkt,  E_DIM);
    transpose_w_bf16<<<tg, tb, 0, stream>>>(Wv,  wvt,  E_DIM);
    transpose_w_bf16<<<tg, tb, 0, stream>>>(Wo,  wot,  E_DIM);
    transpose_w_bf16<<<tg, tb, 0, stream>>>(Wln, wlnt, E_DIM);

    // 32x64 tile per wave: (4096/32)*(1024/64) = 2048 waves -> 256 blocks
    const int gemm_blocks = (ROWS / 32) * (E_DIM / 64) / 8;
    // Q/K/V projections (Q pre-scaled by Dh^-0.5 = 0.125)
    gemm_wmma<<<gemm_blocks, 256, 0, stream>>>(xb, E_DIM, wqt, bq, nullptr,
                                               qhb, nullptr, ROWS, E_DIM, E_DIM, 0.125f, 1);
    gemm_wmma<<<gemm_blocks, 256, 0, stream>>>(xb, E_DIM, wkt, bk, nullptr,
                                               khb, nullptr, ROWS, E_DIM, E_DIM, 1.0f, 1);
    gemm_wmma<<<gemm_blocks, 256, 0, stream>>>(xb, E_DIM, wvt, bv, nullptr,
                                               vtb, nullptr, ROWS, E_DIM, E_DIM, 1.0f, 3);

    // attention: 32 heads * 128 query tiles = 4096 waves
    attn_fwd<<<512, 256, 0, stream>>>(qhb, khb, vtb, attnb);

    // out-proj with fused residual add (fp32 x), bf16 result
    gemm_wmma<<<gemm_blocks, 256, 0, stream>>>(attnb, E_DIM, wot, bo, x,
                                               yb, nullptr, ROWS, E_DIM, E_DIM, 1.0f, 0);
    // final matmul -> fp32 output
    gemm_wmma<<<gemm_blocks, 256, 0, stream>>>(yb, E_DIM, wlnt, bln, nullptr,
                                               nullptr, out, ROWS, E_DIM, E_DIM, 1.0f, 4);
}
